// MultiHeadAttention_46952582480383
// MI455X (gfx1250) — compile-verified, matrix operand order checked by reading, not run on hardware
//
#include <hip/hip_runtime.h>
#include <stdint.h>

// ---------------------------------------------------------------------------
// MHA with QKV projections for MI455X (gfx1250, wave32, WMMA 16x16x32 bf16).
// Kernel 1: fused QKV projection GEMMs (f32 in, bf16 out, f32 accumulate).
// Kernel 2: flash attention (never materializes the S x S score matrix).
// All stores/loads use one per-lane base pointer + immediate offsets.
// ---------------------------------------------------------------------------

typedef __bf16 bf16_t;
typedef __attribute__((ext_vector_type(4)))  __bf16 v4bf;
typedef __attribute__((ext_vector_type(8)))  __bf16 v8bf;
typedef __attribute__((ext_vector_type(16))) __bf16 v16bf;
typedef __attribute__((ext_vector_type(4)))  float  v4f;
typedef __attribute__((ext_vector_type(8)))  float  v8f;

#define B_   4
#define S_   2048
#define E_   1024
#define H_   16
#define D_   64
#define HD_  1024
#define M_   (B_ * S_)          // 8192 flattened (b,s) rows

static __device__ __forceinline__ v16bf cat8(v8bf a, v8bf b) {
  return __builtin_shufflevector(a, b, 0,1,2,3,4,5,6,7,8,9,10,11,12,13,14,15);
}
static __device__ __forceinline__ v8f wmma_bf16(v16bf a, v16bf b, v8f c) {
  // emits v_wmma_f32_16x16x32_bf16
  return __builtin_amdgcn_wmma_f32_16x16x32_bf16(false, a, false, b,
                                                 (short)0, c, false, false);
}
static __device__ __forceinline__ v8f vzero8() {
  v8f z = {0.f, 0.f, 0.f, 0.f, 0.f, 0.f, 0.f, 0.f};
  return z;
}

// ---------------------------------------------------------------------------
// Kernel 1: Y = X @ W^T  (M=8192, N=1024, K=1024), blockIdx.z picks Q/K/V.
// Block = 256 threads (8 waves), tile 128x128, K-step 32.
// LDS rows padded to 40 bf16 (80B): per-lane ds_load_b128 fragment reads hit
// start-banks {20*r mod 64} which tile the 64 banks perfectly (no conflicts).
// ---------------------------------------------------------------------------
#define LDST 40

__global__ __launch_bounds__(256)
void qkv_proj_kernel(const float* __restrict__ q, const float* __restrict__ h,
                     const float* __restrict__ Wq, const float* __restrict__ Wk,
                     const float* __restrict__ Wv,
                     bf16_t* __restrict__ Qb, bf16_t* __restrict__ Kb,
                     bf16_t* __restrict__ Vt) {
  __shared__ __align__(16) bf16_t As[128 * LDST];
  __shared__ __align__(16) bf16_t Bs[128 * LDST];

  const int which = blockIdx.z;
  const float* __restrict__ X = (which == 0) ? q : h;
  const float* __restrict__ W = (which == 0) ? Wq : ((which == 1) ? Wk : Wv);

  const int m0   = blockIdx.x * 128;
  const int n0   = blockIdx.y * 128;
  const int tid  = threadIdx.x;
  const int lane = tid & 31;
  const int w    = tid >> 5;     // 0..7
  const int wr   = w >> 1;       // 0..3 : 32-row group
  const int wc   = w & 1;        // 0..1 : 64-col group
  const int ln   = lane & 15;
  const int hi   = lane >> 4;    // lane half selects K chunk {0..7,16..23} vs {8..15,24..31}
  const int koff = hi * 8;

  // LDS fragment base addresses are loop-invariant
  const bf16_t* paA0 = &As[(wr * 32 + ln) * LDST + koff];            // mt offset: 16*LDST
  const bf16_t* pbB0 = &Bs[(wc * 64 + ln) * LDST + koff];            // nt offset: 16*LDST

  v8f acc[2][4];
  for (int mt = 0; mt < 2; ++mt)
    for (int nt = 0; nt < 4; ++nt) acc[mt][nt] = vzero8();

  for (int k0 = 0; k0 < E_; k0 += 32) {
    __syncthreads();
    // stage 128x32 f32 -> bf16 for both A (activations) and B (weights)
#pragma unroll
    for (int p = 0; p < 4; ++p) {
      int id  = p * 256 + tid;
      int row = id >> 3;
      int cc  = (id & 7) * 4;
      v4f a = *(const v4f*)&X[(size_t)(m0 + row) * E_ + k0 + cc];
      v4f b = *(const v4f*)&W[(size_t)(n0 + row) * E_ + k0 + cc];
      v4bf ab = {(bf16_t)a[0], (bf16_t)a[1], (bf16_t)a[2], (bf16_t)a[3]};
      v4bf bb = {(bf16_t)b[0], (bf16_t)b[1], (bf16_t)b[2], (bf16_t)b[3]};
      *(v4bf*)&As[row * LDST + cc] = ab;
      *(v4bf*)&Bs[row * LDST + cc] = bb;
    }
    __syncthreads();

    v16bf afr[2], bfr[4];
#pragma unroll
    for (int mt = 0; mt < 2; ++mt) {
      const bf16_t* pa = paA0 + mt * 16 * LDST;
      afr[mt] = cat8(*(const v8bf*)pa, *(const v8bf*)(pa + 16));
    }
#pragma unroll
    for (int nt = 0; nt < 4; ++nt) {
      const bf16_t* pb = pbB0 + nt * 16 * LDST;
      bfr[nt] = cat8(*(const v8bf*)pb, *(const v8bf*)(pb + 16));
    }
#pragma unroll
    for (int mt = 0; mt < 2; ++mt)
#pragma unroll
      for (int nt = 0; nt < 4; ++nt)
        acc[mt][nt] = wmma_bf16(afr[mt], bfr[nt], acc[mt][nt]);
  }

  // ------------------------------------------------------------------------
  // Epilogue. Wave-uniform: b = m0/S (128-row tile never crosses a batch),
  // head = (n0 + wc*64)/64 (64-col wave slice never crosses a head).
  // C/D layout: VGPR i -> rows i (lanes 0-15) / i+8 (lanes 16-31), col=lane&15.
  // One per-lane base pointer; every store uses an immediate offset.
  // ------------------------------------------------------------------------
  const int b     = m0 >> 11;                 // m0 / S_
  const int s0    = (m0 & (S_ - 1)) + wr * 32 + hi * 8;
  const int head  = (n0 + wc * 64) >> 6;
  const size_t bh = (size_t)(b * H_ + head);

  if (which == 2) {
    // V transposed (B,H,D,S): d = nt*16 + ln, s = s0 + mt*16 + i
    bf16_t* dst = Vt + (bh * D_ + ln) * S_ + s0;
#pragma unroll
    for (int nt = 0; nt < 4; ++nt)
#pragma unroll
      for (int mt = 0; mt < 2; ++mt)
#pragma unroll
        for (int i = 0; i < 8; ++i)
          dst[nt * 16 * S_ + mt * 16 + i] = (bf16_t)acc[mt][nt][i];
  } else {
    // Q/K (B,H,S,D): s = s0 + mt*16 + i, d = nt*16 + ln
    bf16_t* dst = ((which == 0) ? Qb : Kb) + (bh * S_ + s0) * D_ + ln;
#pragma unroll
    for (int mt = 0; mt < 2; ++mt)
#pragma unroll
      for (int i = 0; i < 8; ++i)
#pragma unroll
        for (int nt = 0; nt < 4; ++nt)
          dst[(mt * 16 + i) * D_ + nt * 16] = (bf16_t)acc[mt][nt][i];
  }
}

// ---------------------------------------------------------------------------
// Kernel 2: flash attention. Block = 128 threads (4 waves), wave owns 16
// queries; streams keys in blocks of 32. K/V fragments come straight from
// global (per-(b,h) K+V = 512 KB -> L2 resident; 192 MB L2).
// ---------------------------------------------------------------------------
#define PST 40

__global__ __launch_bounds__(128)
void flash_attn_kernel(const bf16_t* __restrict__ Qb, const bf16_t* __restrict__ Kb,
                       const bf16_t* __restrict__ Vt,
                       const unsigned char* __restrict__ mask,
                       float* __restrict__ out) {
  __shared__ __align__(16) bf16_t Ps[4][16 * PST];   // per-wave P transpose staging

  const int tid  = threadIdx.x;
  const int w    = tid >> 5;
  const int lane = tid & 31;
  const int ln   = lane & 15;
  const int hi   = lane >> 4;
  const int koff = hi * 8;

  const int b  = blockIdx.z;
  const int hh = blockIdx.y;
  const int q0 = blockIdx.x * 64 + w * 16;

  const size_t bh = (size_t)(b * H_ + hh);
  const bf16_t* Qp = Qb + bh * S_ * D_;

  // Q A-fragments: 16 queries x 64 d -> two WMMA K-steps of 32
  v16bf qf[2];
  {
    const bf16_t* p = Qp + (size_t)(q0 + ln) * D_ + koff;
    qf[0] = cat8(*(const v8bf*)(p),      *(const v8bf*)(p + 16));
    qf[1] = cat8(*(const v8bf*)(p + 32), *(const v8bf*)(p + 48));
  }

  // Per-lane streaming base pointers (advanced by a fixed stride per block;
  // all nt/ks/dt/i variation is immediate offsets)
  const bf16_t*        kptr = Kb + bh * S_ * D_ + (size_t)ln * D_ + koff;
  const bf16_t*        vptr = Vt + bh * D_ * S_ + (size_t)ln * S_ + koff;
  const unsigned char* mptr = mask + (size_t)b * S_ * S_ + (size_t)(q0 + hi * 8) * S_ + ln;

  // Loop-invariant LDS staging addresses
  bf16_t*       ps_st = &Ps[w][hi * 8 * PST + ln];     // store: + i*PST (+16 for nt=1)
  const bf16_t* ps_ld = &Ps[w][ln * PST + koff];       // reload as A-fragment

  float mrow[8], lrow[8];
  v8f acc[4];
#pragma unroll
  for (int i = 0; i < 8; ++i) { mrow[i] = -__builtin_inff(); lrow[i] = 0.f; }
#pragma unroll
  for (int dt = 0; dt < 4; ++dt) acc[dt] = vzero8();

  for (int kb = 0; kb < S_ / 32; ++kb) {
    if (kb + 1 < S_ / 32) {  // global_prefetch_b8 of next K/V block
      __builtin_prefetch(kptr + 32 * D_, 0, 0);
      __builtin_prefetch(vptr + 32, 0, 0);
    }

    // S = Q K^T over two 16-key tiles (K B-fragment: lane = key column)
    v8f s[2];
#pragma unroll
    for (int nt = 0; nt < 2; ++nt) {
      const bf16_t* p = kptr + nt * 16 * D_;
      v16bf kf0 = cat8(*(const v8bf*)(p),      *(const v8bf*)(p + 16));
      v16bf kf1 = cat8(*(const v8bf*)(p + 32), *(const v8bf*)(p + 48));
      v8f sa = vzero8();
      sa = wmma_bf16(qf[0], kf0, sa);
      sa = wmma_bf16(qf[1], kf1, sa);
      s[nt] = sa;
    }

    // scale (1/sqrt(64)) + mask (mask==1 -> disallowed)
#pragma unroll
    for (int nt = 0; nt < 2; ++nt)
#pragma unroll
      for (int i = 0; i < 8; ++i) {
        float v = s[nt][i] * 0.125f;
        s[nt][i] = mptr[i * S_ + nt * 16] ? -1e30f : v;
      }

    // online softmax; each row lives in a 16-lane half -> xor-shuffle reduce
#pragma unroll
    for (int i = 0; i < 8; ++i) {
      float rm = fmaxf(s[0][i], s[1][i]);
      rm = fmaxf(rm, __shfl_xor(rm, 1, 32));
      rm = fmaxf(rm, __shfl_xor(rm, 2, 32));
      rm = fmaxf(rm, __shfl_xor(rm, 4, 32));
      rm = fmaxf(rm, __shfl_xor(rm, 8, 32));
      float nm    = fmaxf(mrow[i], rm);
      float alpha = __expf(mrow[i] - nm);
      float p0    = __expf(s[0][i] - nm);
      float p1    = __expf(s[1][i] - nm);
      float rs = p0 + p1;
      rs += __shfl_xor(rs, 1, 32);
      rs += __shfl_xor(rs, 2, 32);
      rs += __shfl_xor(rs, 4, 32);
      rs += __shfl_xor(rs, 8, 32);
      lrow[i] = lrow[i] * alpha + rs;
      mrow[i] = nm;
#pragma unroll
      for (int dt = 0; dt < 4; ++dt) acc[dt][i] *= alpha;
      ps_st[i * PST]      = (bf16_t)p0;
      ps_st[i * PST + 16] = (bf16_t)p1;
    }

    // reload P as A-fragment (16 q x 32 keys); same-wave LDS RAW is in-order
    v16bf pf = cat8(*(const v8bf*)ps_ld, *(const v8bf*)(ps_ld + 16));

    // O += P V  (V transposed layout makes per-lane chunks contiguous)
#pragma unroll
    for (int dt = 0; dt < 4; ++dt) {
      const bf16_t* p = vptr + dt * 16 * S_;
      v16bf vf = cat8(*(const v8bf*)p, *(const v8bf*)(p + 16));
      acc[dt] = wmma_bf16(pf, vf, acc[dt]);
    }

    kptr += 32 * D_;
    vptr += 32;
    mptr += 32;
  }

  // normalize and store (B, S, H*D) f32; single base + immediate offsets
  float* op = out + ((size_t)(b * S_ + q0 + hi * 8)) * HD_ + hh * D_ + ln;
#pragma unroll
  for (int i = 0; i < 8; ++i) {
    float inv = 1.0f / lrow[i];
#pragma unroll
    for (int dt = 0; dt < 4; ++dt)
      op[i * HD_ + dt * 16] = acc[dt][i] * inv;
  }
}

// ---------------------------------------------------------------------------
extern "C" void kernel_launch(void* const* d_in, const int* in_sizes, int n_in,
                              void* d_out, int out_size, void* d_ws, size_t ws_size,
                              hipStream_t stream) {
  (void)in_sizes; (void)n_in; (void)out_size; (void)ws_size;

  const float*         q    = (const float*)d_in[0];
  const float*         h    = (const float*)d_in[1];
  const unsigned char* mask = (const unsigned char*)d_in[2];
  const float*         Wq   = (const float*)d_in[3];
  const float*         Wk   = (const float*)d_in[4];
  const float*         Wv   = (const float*)d_in[5];
  float*               out  = (float*)d_out;

  const size_t nQKV = (size_t)B_ * H_ * S_ * D_;   // 8.4M bf16 each
  bf16_t* Qb = (bf16_t*)d_ws;
  bf16_t* Kb = Qb + nQKV;
  bf16_t* Vt = Kb + nQKV;                          // 48 MB total workspace

  dim3 gproj(M_ / 128, HD_ / 128, 3);
  qkv_proj_kernel<<<gproj, dim3(256, 1, 1), 0, stream>>>(q, h, Wq, Wk, Wv,
                                                         Qb, Kb, Vt);

  dim3 gattn(S_ / 64, H_, B_);
  flash_attn_kernel<<<gattn, dim3(128, 1, 1), 0, stream>>>(Qb, Kb, Vt, mask, out);
}
